// SwinMamba_59064390255113
// MI455X (gfx1250) — matrix operand order repaired and checked by reading, not compiled, not run on hardware
//
#include <hip/hip_runtime.h>

typedef __attribute__((ext_vector_type(16))) _Float16 v16h;
typedef __attribute__((ext_vector_type(8)))  _Float16 v8h;
typedef __attribute__((ext_vector_type(4)))  _Float16 v4h;
typedef __attribute__((ext_vector_type(8)))  float    v8f;

// ---------------------------------------------------------------- patch embed
// x: (32,3,224,224) NCHW f32 ; w: (64,3,4,4) OIHW ; y: (32,56,56,64) NHWC f32
__global__ __launch_bounds__(256) void patch_embed_k(const float* __restrict__ x,
    const float* __restrict__ w, const float* __restrict__ b, float* __restrict__ y)
{
  size_t i = (size_t)blockIdx.x * 256 + threadIdx.x;
  const size_t total = (size_t)32 * 56 * 56 * 64;
  if (i >= total) return;
  int co = (int)(i & 63);
  size_t r = i >> 6;
  int ow = (int)(r % 56); r /= 56;
  int oh = (int)(r % 56); r /= 56;
  int bb = (int)r;
  float acc = b[co];
  for (int ci = 0; ci < 3; ++ci)
    for (int kh = 0; kh < 4; ++kh)
      for (int kw = 0; kw < 4; ++kw) {
        float xv = x[(((size_t)bb * 3 + ci) * 224 + (oh * 4 + kh)) * 224 + (ow * 4 + kw)];
        float wv = w[((co * 3 + ci) * 4 + kh) * 4 + kw];
        acc += xv * wv;
      }
  y[i] = acc;
}

// ---------------------------------------------------------------- window mamba
template<int C>
__global__ __launch_bounds__(256) void win_mamba_k(const float* __restrict__ X, float* __restrict__ Y,
    const float* __restrict__ n1g, const float* __restrict__ n1b,
    const float* __restrict__ Ap, const float* __restrict__ Bp,
    const float* __restrict__ Cp, const float* __restrict__ Dp,
    const float* __restrict__ lng, const float* __restrict__ lnb,
    int H, int W, int shift)
{
  __shared__ float    win[49 * C];
  __shared__ _Float16 yb [49 * C];
  const int tid = threadIdx.x, lane = tid & 31, wid = tid >> 5;
  const int Wn = W / 7;
  int widx = blockIdx.x;
  int wc = widx % Wn; widx /= Wn;
  int wr = widx % (H / 7);
  int b  = widx / (H / 7);
  const int sh = shift ? 3 : 0;

  for (int t = wid; t < 49; t += 8) {
    int r = t / 7, c2 = t - r * 7;
    int gy = wr * 7 + r + sh;  if (gy >= H) gy -= H;
    int gx = wc * 7 + c2 + sh; if (gx >= W) gx -= W;
    const float* src = X + (((size_t)b * H + gy) * W + gx) * C;
    float s = 0.f, ss = 0.f;
    for (int c = lane; c < C; c += 32) { float v = src[c]; win[t * C + c] = v; s += v; ss += v * v; }
    for (int o = 16; o > 0; o >>= 1) { s += __shfl_xor(s, o, 32); ss += __shfl_xor(ss, o, 32); }
    float m = s * (1.f / C), inv = rsqrtf(ss * (1.f / C) - m * m + 1e-5f);
    for (int c = lane; c < C; c += 32) win[t * C + c] = (win[t * C + c] - m) * inv * n1g[c] + n1b[c];
  }
  __syncthreads();

  for (int dir = 0; dir < 4; ++dir) {
    for (int c = tid; c < C; c += 256) {
      float a = Ap[c], bB = Bp[c], cC = Cp[c], dD = Dp[c];
      float s = 0.f;
      for (int t = 0; t < 49; ++t) {
        int r = t / 7, cl = t - r * 7;
        int pt = (dir == 0) ? t
               : (dir == 1) ? 48 - t
               : (dir == 2) ? r * 7 + (6 - cl)
                            : (6 - r) * 7 + cl;
        float xv = win[pt * C + c];
        s = a * s + bB * xv;
        yb[t * C + c] = (_Float16)(cC * s + dD * xv);
      }
    }
    __syncthreads();
    for (int t = wid; t < 49; t += 8) {
      float s = 0.f, ss = 0.f;
      for (int c = lane; c < C; c += 32) { float v = (float)yb[t * C + c]; s += v; ss += v * v; }
      for (int o = 16; o > 0; o >>= 1) { s += __shfl_xor(s, o, 32); ss += __shfl_xor(ss, o, 32); }
      float m = s * (1.f / C), inv = rsqrtf(ss * (1.f / C) - m * m + 1e-5f);
      int r = t / 7, c2 = t - r * 7;
      int gy = wr * 7 + r + sh;  if (gy >= H) gy -= H;
      int gx = wc * 7 + c2 + sh; if (gx >= W) gx -= W;
      float* dst = Y + (((size_t)b * H + gy) * W + gx) * C;
      for (int c = lane; c < C; c += 32) {
        float v = (((float)yb[t * C + c] - m) * inv * lng[c] + lnb[c]) * 0.25f;
        if (dir == 0) dst[c] = v; else dst[c] += v;
      }
    }
    __syncthreads();
  }
}

// ---------------------------------------------------------------- LN rows -> f16
__global__ __launch_bounds__(256) void ln_rows_k(const float* __restrict__ X,
    const float* __restrict__ g, const float* __restrict__ bta,
    _Float16* __restrict__ Y, int T, int C)
{
  int wid = threadIdx.x >> 5, lane = threadIdx.x & 31;
  int row = blockIdx.x * 8 + wid;
  if (row >= T) return;
  const float* xr = X + (size_t)row * C;
  float s = 0.f, ss = 0.f;
  for (int c = lane; c < C; c += 32) { float v = xr[c]; s += v; ss += v * v; }
  for (int o = 16; o > 0; o >>= 1) { s += __shfl_xor(s, o, 32); ss += __shfl_xor(ss, o, 32); }
  float m = s * (1.f / C), inv = rsqrtf(ss * (1.f / C) - m * m + 1e-5f);
  _Float16* yr = Y + (size_t)row * C;
  for (int c = lane; c < C; c += 32) yr[c] = (_Float16)((xr[c] - m) * inv * g[c] + bta[c]);
}

// ---------------------------------------------------------------- f32 -> f16 transpose
// bt (N x K) = a (K x N) transposed; consecutive output idx walks K (coalesced writes)
__global__ __launch_bounds__(256) void transpose_f16_k(const float* __restrict__ a,
    _Float16* __restrict__ bt, int K, int N)
{
  int i = blockIdx.x * 256 + threadIdx.x;
  if (i >= K * N) return;
  int k = i % K, n = i / K;
  bt[i] = (_Float16)a[(size_t)k * N + n];
}

// ---------------------------------------------------------------- merge gather
__global__ __launch_bounds__(256) void merge_gather_k(const float* __restrict__ X,
    _Float16* __restrict__ Y, int H, int W, int C)
{
  int H2 = H / 2, W2 = W / 2;
  size_t total = (size_t)32 * H2 * W2 * 4 * C;
  size_t i = (size_t)blockIdx.x * 256 + threadIdx.x;
  if (i >= total) return;
  int c = (int)(i % C); size_t t = i / C;
  int q  = (int)(t % 4); t /= 4;
  int ow = (int)(t % W2); t /= W2;
  int oh = (int)(t % H2);
  int b  = (int)(t / H2);
  int dr = q & 1, dc = q >> 1;
  Y[i] = (_Float16)X[(((size_t)b * H + 2 * oh + dr) * W + 2 * ow + dc) * C + c];
}

// ---------------------------------------------------------------- head pool+LN -> f16
__global__ __launch_bounds__(256) void pool_ln_k(const float* __restrict__ X,
    const float* __restrict__ g, const float* __restrict__ bt, _Float16* __restrict__ Y)
{
  __shared__ float pool[512];
  __shared__ float red[2][8];
  int b = blockIdx.x, tid = threadIdx.x, lane = tid & 31, wid = tid >> 5;
  for (int c = tid; c < 512; c += 256) {
    float s = 0.f;
    for (int t = 0; t < 49; ++t) s += X[((size_t)b * 49 + t) * 512 + c];
    pool[c] = s * (1.f / 49.f);
  }
  __syncthreads();
  float s = 0.f, ss = 0.f;
  for (int c = tid; c < 512; c += 256) { float v = pool[c]; s += v; ss += v * v; }
  for (int o = 16; o > 0; o >>= 1) { s += __shfl_xor(s, o, 32); ss += __shfl_xor(ss, o, 32); }
  if (lane == 0) { red[0][wid] = s; red[1][wid] = ss; }
  __syncthreads();
  if (tid == 0) {
    float a = 0.f, aa = 0.f;
    for (int w = 0; w < 8; ++w) { a += red[0][w]; aa += red[1][w]; }
    float m = a * (1.f / 512.f);
    red[0][0] = m;
    red[1][0] = rsqrtf(aa * (1.f / 512.f) - m * m + 1e-5f);
  }
  __syncthreads();
  float m = red[0][0], inv = red[1][0];
  for (int c = tid; c < 512; c += 256)
    Y[(size_t)b * 512 + c] = (_Float16)((pool[c] - m) * inv * g[c] + bt[c]);
}

// ---------------------------------------------------------------- WMMA GEMM
// C_out = A(MxK,f16,row-major) * Bt(NxK,f16,row-major == B(KxN) col-major) + bias.
// Tiles moved to LDS with CDNA5 async-to-LDS (ASYNCcnt), fragments read as b128 runs.
// MODE 0: out f16 = gelu(acc+bias) ; MODE 1: out f32 = acc+bias+res ; MODE 2: out f32 = acc+bias
// Requires M%32==0, N%64==0, K%32==0 (true for every GEMM in this net).
template<int MODE>
__global__ __launch_bounds__(256) void gemm_wmma_k(const _Float16* __restrict__ A,
    const _Float16* __restrict__ Bt, const float* __restrict__ bias,
    const float* __restrict__ res, void* __restrict__ out, int M, int N, int K)
{
  __shared__ _Float16 As[32 * 32];   // [m][k]
  __shared__ _Float16 Bs[64 * 32];   // [n][k]
  const int tid = threadIdx.x, lane = tid & 31, wid = tid >> 5;
  const int tileM = blockIdx.y * 32, tileN = blockIdx.x * 64;
  const int wM = (wid >> 2) * 16;    // 0 or 16
  const int wN = (wid & 3) * 16;     // 0,16,32,48

  // per-thread tile-copy coordinates
  const int ar = tid >> 3, ac = (tid & 7) * 4;      // A: 32x32, 4 halfs (8B) per thread
  const int bn = tid >> 2, bk = (tid & 3) * 8;      // B: 64x32, 8 halfs (16B) per thread
  const unsigned lAs = (unsigned)(size_t)&As[ar * 32 + ac];
  const unsigned lBs = (unsigned)(size_t)&Bs[bn * 32 + bk];

  v8f acc = {};
  for (int k0 = 0; k0 < K; k0 += 32) {
    const _Float16* ga = A  + (size_t)(tileM + ar) * K + k0 + ac;
    const _Float16* gb = Bt + (size_t)(tileN + bn) * K + k0 + bk;
    // CDNA5 async DMA: global -> LDS without VGPR staging (tracked by ASYNCcnt)
    asm volatile("global_load_async_to_lds_b64 %0, %1, off"
                 :: "v"(lAs), "v"(ga) : "memory");
    asm volatile("global_load_async_to_lds_b128 %0, %1, off"
                 :: "v"(lBs), "v"(gb) : "memory");
    if (k0 + 32 < K) {            // prefetch next K-step into L2/WGP$
      __builtin_prefetch(ga + 32, 0, 1);
      __builtin_prefetch(gb + 32, 0, 1);
    }
    asm volatile("s_wait_asynccnt 0" ::: "memory");
    __syncthreads();

    // A fragment: 16-bit A 16x32 layout -> two contiguous 8-half runs
    const int m   = (lane & 15) + wM;
    const int kbA = (lane >> 4) * 8;
    v8h a0 = *(const v8h*)&As[m * 32 + kbA];
    v8h a1 = *(const v8h*)&As[m * 32 + kbA + 16];
    v16h af = __builtin_shufflevector(a0, a1, 0,1,2,3,4,5,6,7,8,9,10,11,12,13,14,15);
    // B fragment: column per lane, K contiguous -> two contiguous 8-half runs
    const int n   = (lane & 15) + wN;
    const int kbB = (lane >> 4) * 8;
    v8h b0 = *(const v8h*)&Bs[n * 32 + kbB];
    v8h b1 = *(const v8h*)&Bs[n * 32 + kbB + 16];
    v16h bf = __builtin_shufflevector(b0, b1, 0,1,2,3,4,5,6,7,8,9,10,11,12,13,14,15);

    acc = __builtin_amdgcn_wmma_f32_16x16x32_f16(false, af, false, bf, (short)0, acc, false, false);
    __syncthreads();
  }

  const int n_g = tileN + wN + (lane & 15);
  const float bv = bias[n_g];
#pragma unroll
  for (int r = 0; r < 8; ++r) {
    const int m_g = tileM + wM + r + ((lane >> 4) << 3);
    float v = acc[r] + bv;
    if (MODE == 0) {
      float gl = 0.5f * v * (1.0f + erff(v * 0.70710678118f));
      ((_Float16*)out)[(size_t)m_g * N + n_g] = (_Float16)gl;
    } else if (MODE == 1) {
      ((float*)out)[(size_t)m_g * N + n_g] = v + res[(size_t)m_g * N + n_g];
    } else {
      ((float*)out)[(size_t)m_g * N + n_g] = v;
    }
  }
}

// ---------------------------------------------------------------- host orchestration
extern "C" void kernel_launch(void* const* d_in, const int* in_sizes, int n_in,
                              void* d_out, int out_size, void* d_ws, size_t ws_size,
                              hipStream_t stream)
{
  (void)out_size; (void)ws_size;
  int pb; const float* xin;
  if (in_sizes[0] == 32 * 3 * 224 * 224) { xin = (const float*)d_in[0]; pb = 1; }
  else { xin = (const float*)d_in[n_in - 1]; pb = 0; }
  auto P = [&](int i) { return (const float*)d_in[pb + i]; };

  struct Blk { const float *b1,*b2,*n1b,*n1g,*n2b,*n2g,*A,*B,*C,*D,*lnb,*lng,*w1,*w2; } blk[12];
  const float *head_b, *head_ln_b, *head_ln_g, *head_w, *patch_b, *patch_w;
  const float *mw[3], *mb[3];

  bool sorted = (in_sizes[pb] == 512);  // sorted-key pytree puts head_b (512) first
  if (sorted) {
    head_b = P(0); head_ln_b = P(1); head_ln_g = P(2); head_w = P(3);
    for (int q = 0; q < 3; ++q) { mb[q] = P(4 + 2 * q); mw[q] = P(5 + 2 * q); }
    patch_b = P(10); patch_w = P(11);
    for (int k = 0; k < 12; ++k) {
      int o = 12 + 14 * k;  // b1,b2,n1_b,n1_g,n2_b,n2_g,A,B,C,D,ln_b,ln_g,w1,w2
      blk[k] = { P(o+0),P(o+1),P(o+2),P(o+3),P(o+4),P(o+5),P(o+6),
                 P(o+7),P(o+8),P(o+9),P(o+10),P(o+11),P(o+12),P(o+13) };
    }
  } else {  // insertion-order flatten
    patch_w = P(0); patch_b = P(1);
    for (int k = 0; k < 12; ++k) {
      int o = 2 + 14 * k;  // n1_g,n1_b,A,B,C,D,ln_g,ln_b,n2_g,n2_b,w1,b1,w2,b2
      blk[k].n1g = P(o+0);  blk[k].n1b = P(o+1);
      blk[k].A = P(o+2); blk[k].B = P(o+3); blk[k].C = P(o+4); blk[k].D = P(o+5);
      blk[k].lng = P(o+6); blk[k].lnb = P(o+7);
      blk[k].n2g = P(o+8); blk[k].n2b = P(o+9);
      blk[k].w1 = P(o+10); blk[k].b1 = P(o+11); blk[k].w2 = P(o+12); blk[k].b2 = P(o+13);
    }
    for (int q = 0; q < 3; ++q) { mw[q] = P(170 + 2 * q); mb[q] = P(171 + 2 * q); }
    head_ln_g = P(176); head_ln_b = P(177); head_w = P(178); head_b = P(179);
  }

  char* ws = (char*)d_ws;
  float*    act0 = (float*)ws;                       // 25,690,112 B
  float*    act1 = (float*)(ws + 25690112);          // 25,690,112 B
  _Float16* xn   = (_Float16*)(ws + 51380224);       // 12,845,056 B
  _Float16* h1   = (_Float16*)(ws + 64225280);       // 51,380,224 B
  _Float16* w1h  = (_Float16*)(ws + 115605504);      //  2,097,152 B
  _Float16* w2h  = (_Float16*)(ws + 117702656);      //  2,097,152 B
  _Float16* hln  = (_Float16*)(ws + 119799808);      //     32,768 B

  auto gemm = [&](int mode, const _Float16* Am, const _Float16* Bm, const float* bias,
                  const float* res, void* outp, int M, int N, int K) {
    dim3 g(N / 64, M / 32), t(256);
    if (mode == 0)      gemm_wmma_k<0><<<g, t, 0, stream>>>(Am, Bm, bias, res, outp, M, N, K);
    else if (mode == 1) gemm_wmma_k<1><<<g, t, 0, stream>>>(Am, Bm, bias, res, outp, M, N, K);
    else                gemm_wmma_k<2><<<g, t, 0, stream>>>(Am, Bm, bias, res, outp, M, N, K);
  };

  {
    size_t total = (size_t)32 * 56 * 56 * 64;
    patch_embed_k<<<dim3((unsigned)((total + 255) / 256)), dim3(256), 0, stream>>>(
        xin, patch_w, patch_b, act0);
  }

  const int dims[4]   = { 64, 128, 256, 512 };
  const int depths[4] = { 2, 2, 6, 2 };
  int H = 56, W = 56;
  float* cur = act0;
  float* alt = act1;
  int bki = 0;
  for (int si = 0; si < 4; ++si) {
    const int d = dims[si];
    const int T = 32 * H * W;
    for (int bi = 0; bi < depths[si]; ++bi, ++bki) {
      const Blk& bp = blk[bki];
      const int Nw = 32 * (H / 7) * (W / 7);
      const int shift = (bi % 2 == 1) ? 1 : 0;
      switch (d) {
        case 64:  win_mamba_k<64 ><<<Nw, 256, 0, stream>>>(cur, alt, bp.n1g, bp.n1b, bp.A, bp.B, bp.C, bp.D, bp.lng, bp.lnb, H, W, shift); break;
        case 128: win_mamba_k<128><<<Nw, 256, 0, stream>>>(cur, alt, bp.n1g, bp.n1b, bp.A, bp.B, bp.C, bp.D, bp.lng, bp.lnb, H, W, shift); break;
        case 256: win_mamba_k<256><<<Nw, 256, 0, stream>>>(cur, alt, bp.n1g, bp.n1b, bp.A, bp.B, bp.C, bp.D, bp.lng, bp.lnb, H, W, shift); break;
        default:  win_mamba_k<512><<<Nw, 256, 0, stream>>>(cur, alt, bp.n1g, bp.n1b, bp.A, bp.B, bp.C, bp.D, bp.lng, bp.lnb, H, W, shift); break;
      }
      ln_rows_k<<<T / 8, 256, 0, stream>>>(alt, bp.n2g, bp.n2b, xn, T, d);
      // weights transposed to NxK for contiguous B fragments
      transpose_f16_k<<<(d * 4 * d + 255) / 256, 256, 0, stream>>>(bp.w1, w1h, d, 4 * d);
      transpose_f16_k<<<(4 * d * d + 255) / 256, 256, 0, stream>>>(bp.w2, w2h, 4 * d, d);
      gemm(0, xn, w1h, bp.b1, nullptr, h1, T, 4 * d, d);
      gemm(1, h1, w2h, bp.b2, alt, cur, T, d, 4 * d);
    }
    if (si < 3) {
      const int H2 = H / 2, W2 = W / 2, T2 = 32 * H2 * W2;
      size_t tot = (size_t)T2 * 4 * d;
      merge_gather_k<<<dim3((unsigned)((tot + 255) / 256)), dim3(256), 0, stream>>>(cur, xn, H, W, d);
      transpose_f16_k<<<(4 * d * 2 * d + 255) / 256, 256, 0, stream>>>(mw[si], w1h, 4 * d, 2 * d);
      gemm(2, xn, w1h, mb[si], nullptr, alt, T2, 2 * d, 4 * d);
      float* t = cur; cur = alt; alt = t;
      H = H2; W = W2;
    }
  }

  pool_ln_k<<<32, 256, 0, stream>>>(cur, head_ln_g, head_ln_b, hln);
  transpose_f16_k<<<(512 * 512 + 255) / 256, 256, 0, stream>>>(head_w, w1h, 512, 512);
  gemm(2, hln, w1h, head_b, nullptr, (float*)d_out, 32, 512, 512);
}
